// GenericCrossAttentionBlock_4552665334362
// MI455X (gfx1250) — matrix operand order repaired and checked by reading, not compiled, not run on hardware
//
#include <hip/hip_runtime.h>

// ---------------------------------------------------------------------------
// MI455X (gfx1250, wave32) cross-attention block.
// f16 WMMA (v_wmma_f32_16x16x32_f16) for all GEMMs + flash attention,
// async LDS staging (global_load_async_to_lds_b128 + s_wait_asynccnt).
// ---------------------------------------------------------------------------

typedef __attribute__((ext_vector_type(16))) _Float16 v16h;
typedef __attribute__((ext_vector_type(8)))  float    v8f;
typedef __attribute__((ext_vector_type(4)))  int      v4i;

#if defined(__has_builtin)
#  if __has_builtin(__builtin_amdgcn_global_load_async_to_lds_b128)
#    define HAVE_ASYNC_COPY 1
#  endif
#  if __has_builtin(__builtin_amdgcn_s_wait_asynccnt)
#    define WAIT_ASYNC(n) __builtin_amdgcn_s_wait_asynccnt(n)
#  endif
#endif
#ifndef WAIT_ASYNC
#  define WAIT_ASYNC(n) asm volatile("s_wait_asynccnt %0" ::"n"(n) : "memory")
#endif

// 16-byte chunk copy global -> LDS. Async DMA path on CDNA5, sync fallback.
// Builtin prototype (from hipcc diagnostic): (int4 AS1*, int4 AS3*, Imm, Imm).
__device__ __forceinline__ void copy16(_Float16* lds, const _Float16* g) {
#ifdef HAVE_ASYNC_COPY
  __builtin_amdgcn_global_load_async_to_lds_b128(
      (__attribute__((address_space(1))) v4i*)g,
      (__attribute__((address_space(3))) v4i*)lds, /*offset=*/0, /*cpol=*/0);
#else
  *reinterpret_cast<uint4*>(lds) = *reinterpret_cast<const uint4*>(g);
#endif
}

__device__ __forceinline__ v8f wmma_f16(v16h a, v16h b, v8f c) {
  return __builtin_amdgcn_wmma_f32_16x16x32_f16(
      /*neg_a=*/false, a, /*neg_b=*/false, b,
      /*c_mod=*/(short)0, c, /*reuse_a=*/false, /*reuse_b=*/false);
}

// A fragment: 16x32 f16 (M x K), source row-major [M][ld].
// Layout: lanes 0-15 -> K 0..15 ; lanes 16-31 -> K 16..31.
__device__ __forceinline__ v16h load_A_frag(const _Float16* p, int ld) {
  const int lane = threadIdx.x & 31;
  const int m = lane & 15;
  const int kh = (lane >> 4) << 4;  // 0 or 16
  const _Float16* row = p + (size_t)m * ld;
  v16h a;
#pragma unroll
  for (int i = 0; i < 16; ++i) a[i] = row[kh + i];
  return a;
}

// B fragment: 32x16 f16 (K x N), source stored [N][ld] (B-transposed / weight
// row-major / key rows). Lanes 0-15: K 0..15; lanes 16-31: K 16..31.
__device__ __forceinline__ v16h load_B_nk(const _Float16* p, int ld) {
  const int lane = threadIdx.x & 31;
  const int n = lane & 15;
  const int k0 = (lane >> 4) << 4;  // 0 or 16
  const _Float16* row = p + (size_t)n * ld + k0;
  v16h b;
#pragma unroll
  for (int i = 0; i < 16; ++i) b[i] = row[i];
  return b;
}

// ---------------------------------------------------------------------------
// 1) f32 -> f16 cast (weights)
// ---------------------------------------------------------------------------
__global__ __launch_bounds__(256) void cast_f32_f16_kernel(
    const float* __restrict__ s, _Float16* __restrict__ d, int n) {
  int i = blockIdx.x * 256 + threadIdx.x;
  if (i < n) d[i] = (_Float16)s[i];
}

// ---------------------------------------------------------------------------
// 2) RMS-norm + cast, one block (256 threads) per row of length D
// ---------------------------------------------------------------------------
__global__ __launch_bounds__(256) void rmsnorm_cast_kernel(
    const float* __restrict__ x, const float* __restrict__ scale,
    _Float16* __restrict__ y, int D) {
  __shared__ float red[8];
  const size_t row = blockIdx.x;
  const float* xr = x + row * D;
  float ss = 0.f;
  for (int i = threadIdx.x; i < D; i += 256) { float v = xr[i]; ss += v * v; }
#pragma unroll
  for (int off = 16; off; off >>= 1) ss += __shfl_xor(ss, off, 32);
  if ((threadIdx.x & 31) == 0) red[threadIdx.x >> 5] = ss;
  __syncthreads();
  if (threadIdx.x == 0) {
    float t = 0.f;
#pragma unroll
    for (int i = 0; i < 8; ++i) t += red[i];
    red[0] = t;
  }
  __syncthreads();
  const float inv = rsqrtf(red[0] / (float)D + 1e-6f);
  _Float16* yr = y + row * D;
  for (int i = threadIdx.x; i < D; i += 256)
    yr[i] = (_Float16)(xr[i] * scale[i] * inv);
}

// ---------------------------------------------------------------------------
// 3) WMMA GEMM:  C[M][N] = A[M][K] * W[N][K]^T  (+ optional skip), f32 out.
//    Block tile 128x64, 8 waves, K-step 64, double-buffered async LDS staging.
// ---------------------------------------------------------------------------
#define GT_M 128
#define GT_N 64
#define GT_K 64
#define GT_LD (GT_K + 8)  // stride 72 halves = 144B (16B-multiple)

__global__ __launch_bounds__(256) void gemm_wmma_kernel(
    const _Float16* __restrict__ A, const _Float16* __restrict__ W,
    float* __restrict__ C, const float* __restrict__ skip,
    int M, int N, int K, int ldc) {
  __shared__ _Float16 sA[2][GT_M][GT_LD];
  __shared__ _Float16 sW[2][GT_N][GT_LD];
  const int m0 = blockIdx.x * GT_M;
  const int n0 = blockIdx.y * GT_N;
  const int tid = threadIdx.x;
  const int wave = tid >> 5;
  const int lane = tid & 31;

  v8f zero = {};
  v8f acc[4] = {zero, zero, zero, zero};

  auto stage = [&](int buf, int kt) {
    // A tile 128x64 halves = 1024 b128 chunks; 4 per thread.
#pragma unroll
    for (int t = 0; t < 4; ++t) {
      int idx = tid + t * 256;
      int row = idx >> 3, c8 = (idx & 7) << 3;
      copy16(&sA[buf][row][c8], &A[(size_t)(m0 + row) * K + kt + c8]);
    }
    // W tile 64x64 halves = 512 chunks; 2 per thread.
#pragma unroll
    for (int t = 0; t < 2; ++t) {
      int idx = tid + t * 256;
      int row = idx >> 3, c8 = (idx & 7) << 3;
      copy16(&sW[buf][row][c8], &W[(size_t)(n0 + row) * K + kt + c8]);
    }
  };

  stage(0, 0);
  for (int kt = 0; kt < K; kt += GT_K) {
    const int buf = (kt >> 6) & 1;
    if (kt + GT_K < K) {
      stage(buf ^ 1, kt + GT_K);  // overlap next tile DMA with compute
      WAIT_ASYNC(6);              // 6 async ops/wave per stage: current done
    } else {
      WAIT_ASYNC(0);
    }
    __syncthreads();

    const v16h af0 = load_A_frag(&sA[buf][wave * 16][0], GT_LD);
    const v16h af1 = load_A_frag(&sA[buf][wave * 16][32], GT_LD);
#pragma unroll
    for (int f = 0; f < 4; ++f) {
      acc[f] = wmma_f16(af0, load_B_nk(&sW[buf][f * 16][0], GT_LD), acc[f]);
      acc[f] = wmma_f16(af1, load_B_nk(&sW[buf][f * 16][32], GT_LD), acc[f]);
    }
    __syncthreads();
  }

  // C/D layout: lane n=lane&15, rows (lane>>4)*8 + r.
  const int cn = lane & 15;
  const int rb = (lane >> 4) << 3;
#pragma unroll
  for (int f = 0; f < 4; ++f) {
#pragma unroll
    for (int r = 0; r < 8; ++r) {
      size_t row = (size_t)(m0 + wave * 16 + rb + r);
      size_t col = (size_t)(n0 + f * 16 + cn);
      float val = acc[f][r];
      if (skip) val += skip[row * ldc + col];
      C[row * ldc + col] = val;
    }
  }
}

// ---------------------------------------------------------------------------
// 4) Per-head cosine-scale + 3-axis RoPE (D6=10, rotary over dims 0..59).
//    f[h][d6] = pi * exp((d6*16 + h) * ln(10)/160)
// ---------------------------------------------------------------------------
__device__ __forceinline__ void cos_rope_head(
    const float* __restrict__ src, float p0, float p1, float p2, int h,
    float hscale, _Float16* __restrict__ out64) {
  float xv[64];
  float ss = 0.f;
#pragma unroll
  for (int i = 0; i < 64; ++i) { xv[i] = src[i]; ss += xv[i] * xv[i]; }
  const float s = sqrtf(hscale) * rsqrtf(ss + 1e-6f);
#pragma unroll
  for (int i = 0; i < 64; ++i) xv[i] *= s;
  const float step = 2.302585092994046f / 160.f;  // ln(10)/160
#pragma unroll
  for (int j = 0; j < 30; ++j) {
    const int a = j / 10, d6 = j % 10;
    const float f = 3.14159265358979f * __expf((float)(d6 * 16 + h) * step);
    const float pa = (a == 0) ? p0 : ((a == 1) ? p1 : p2);
    const float th = pa * f;
    const float c = __cosf(th), sn = __sinf(th);
    out64[j]      = (_Float16)(xv[j] * c - xv[j + 30] * sn);
    out64[j + 30] = (_Float16)(xv[j + 30] * c + xv[j] * sn);
  }
#pragma unroll
  for (int j = 60; j < 64; ++j) out64[j] = (_Float16)xv[j];
}

// One thread per (b, l, h). q f32 [B*L][1024] -> qh f16 [B][16][L][64]
__global__ __launch_bounds__(256) void q_post_kernel(
    const float* __restrict__ q, const float* __restrict__ pos,
    const float* __restrict__ head_scale, _Float16* __restrict__ qh, int L) {
  const size_t idx = (size_t)blockIdx.x * 256 + threadIdx.x;
  const int h = (int)(idx & 15);
  const size_t bl = idx >> 4;  // b*L + l
  const int l = (int)(bl % L);
  const int b = (int)(bl / L);
  _Float16 o[64];
  cos_rope_head(q + bl * 1024 + h * 64, pos[bl * 3], pos[bl * 3 + 1],
                pos[bl * 3 + 2], h, head_scale[h], o);
  _Float16* dst = qh + (((size_t)b * 16 + h) * L + l) * 64;
#pragma unroll
  for (int i = 0; i < 64; ++i) dst[i] = o[i];
}

// One thread per (b, lc, h). kv f32 [B*LC][2048]:
//   k (col h*64)      -> cosine+rope -> kh f16 [B][16][LC][64]
//   v (col 1024+h*64) -> transpose   -> vt f16 [B][16][64][LC]
__global__ __launch_bounds__(256) void kv_post_kernel(
    const float* __restrict__ kv, const float* __restrict__ pos,
    const float* __restrict__ head_scale, _Float16* __restrict__ khd,
    _Float16* __restrict__ vtd, int LC) {
  const size_t idx = (size_t)blockIdx.x * 256 + threadIdx.x;
  const int h = (int)(idx & 15);
  const size_t bl = idx >> 4;  // b*LC + lc
  const int lc = (int)(bl % LC);
  const int b = (int)(bl / LC);
  _Float16 o[64];
  cos_rope_head(kv + bl * 2048 + h * 64, pos[bl * 3], pos[bl * 3 + 1],
                pos[bl * 3 + 2], h, head_scale[h], o);
  _Float16* kdst = khd + (((size_t)b * 16 + h) * LC + lc) * 64;
#pragma unroll
  for (int i = 0; i < 64; ++i) kdst[i] = o[i];
  const float* vrow = kv + bl * 2048 + 1024 + h * 64;
  _Float16* vcol = vtd + (((size_t)b * 16 + h) * 64) * LC + lc;
#pragma unroll
  for (int i = 0; i < 64; ++i) vcol[(size_t)i * LC] = (_Float16)vrow[i];
}

// ---------------------------------------------------------------------------
// 5) Flash attention. 4 waves/block, wave owns 16 q rows, key blocks of 32.
//    K/V blocks staged once per workgroup via async LDS DMA (double-buffered);
//    S = Q·K^T (f32 frags), online softmax via width-16 shuffles, P staged in
//    per-wave LDS as a 16x32 f16 A-frag, O += P·V.
// ---------------------------------------------------------------------------
__global__ __launch_bounds__(128) void flash_attn_kernel(
    const _Float16* __restrict__ qh,  // [B][H][L][64]
    const _Float16* __restrict__ kh,  // [B][H][LC][64]
    const _Float16* __restrict__ vt,  // [B][H][64][LC]
    _Float16* __restrict__ oh,        // [B][L][1024]
    int L, int LC) {
  __shared__ _Float16 sK[2][32][72];  // 32 keys x 64 dims (+8 pad)
  __shared__ _Float16 sV[2][64][40];  // 64 dims x 32 keys (+8 pad)
  __shared__ _Float16 sP[4][16 * 32];
  const int tid = threadIdx.x;
  const int wave = tid >> 5;
  const int lane = tid & 31;
  const int b = blockIdx.z, h = blockIdx.y;
  const int H = gridDim.y;
  const int q0 = blockIdx.x * 64 + wave * 16;

  const _Float16* qbase = qh + (((size_t)b * H + h) * L + q0) * 64;
  const v16h qa0 = load_A_frag(qbase, 64);
  const v16h qa1 = load_A_frag(qbase + 32, 64);

  v8f zero = {};
  v8f o[4] = {zero, zero, zero, zero};
  float mrow[8], lrow[8];
#pragma unroll
  for (int r = 0; r < 8; ++r) { mrow[r] = -1e30f; lrow[r] = 0.f; }

  const _Float16* kbase = kh + (((size_t)b * H + h) * LC) * 64;
  const _Float16* vbase = vt + (((size_t)b * H + h) * 64) * LC;
  const int cn = lane & 15;
  const int rb = (lane >> 4) << 3;
  _Float16* pw = sP[wave];

  auto stage = [&](int buf, int kb) {
    // K block: 32 rows x 64 halves = 256 chunks; 2 per thread.
#pragma unroll
    for (int t = 0; t < 2; ++t) {
      int idx = tid + t * 128;
      int row = idx >> 3, c8 = (idx & 7) << 3;
      copy16(&sK[buf][row][c8], kbase + (size_t)(kb + row) * 64 + c8);
    }
    // V block: 64 rows x 32 halves = 256 chunks; 2 per thread.
#pragma unroll
    for (int t = 0; t < 2; ++t) {
      int idx = tid + t * 128;
      int row = idx >> 2, c8 = (idx & 3) << 3;
      copy16(&sV[buf][row][c8], vbase + (size_t)row * LC + kb + c8);
    }
  };

  stage(0, 0);
  for (int kb = 0; kb < LC; kb += 32) {
    const int buf = (kb >> 5) & 1;
    if (kb + 32 < LC) {
      stage(buf ^ 1, kb + 32);  // overlap next K/V DMA with this block
      WAIT_ASYNC(4);            // 4 async ops/wave per stage
    } else {
      WAIT_ASYNC(0);
    }
    __syncthreads();

    v8f s0 = zero, s1 = zero;
    s0 = wmma_f16(qa0, load_B_nk(&sK[buf][0][0], 72), s0);
    s0 = wmma_f16(qa1, load_B_nk(&sK[buf][0][32], 72), s0);
    s1 = wmma_f16(qa0, load_B_nk(&sK[buf][16][0], 72), s1);
    s1 = wmma_f16(qa1, load_B_nk(&sK[buf][16][32], 72), s1);

    float p0f[8], p1f[8];
#pragma unroll
    for (int r = 0; r < 8; ++r) {
      float t = fmaxf(s0[r], s1[r]);
      t = fmaxf(t, __shfl_xor(t, 1, 16));
      t = fmaxf(t, __shfl_xor(t, 2, 16));
      t = fmaxf(t, __shfl_xor(t, 4, 16));
      t = fmaxf(t, __shfl_xor(t, 8, 16));
      const float mn = fmaxf(mrow[r], t);
      const float corr = __expf(mrow[r] - mn);
      mrow[r] = mn;
      lrow[r] *= corr;
#pragma unroll
      for (int f = 0; f < 4; ++f) o[f][r] *= corr;
      p0f[r] = __expf(s0[r] - mn);
      p1f[r] = __expf(s1[r] - mn);
      float rs = p0f[r] + p1f[r];
      rs += __shfl_xor(rs, 1, 16);
      rs += __shfl_xor(rs, 2, 16);
      rs += __shfl_xor(rs, 4, 16);
      rs += __shfl_xor(rs, 8, 16);
      lrow[r] += rs;
    }

    // P (16 q-rows x 32 keys) -> per-wave LDS, reload as A-fragment.
#pragma unroll
    for (int r = 0; r < 8; ++r) {
      pw[(rb + r) * 32 + cn]      = (_Float16)p0f[r];
      pw[(rb + r) * 32 + 16 + cn] = (_Float16)p1f[r];
    }
    const v16h pa = load_A_frag(pw, 32);  // same-wave DS dep -> s_wait_dscnt
#pragma unroll
    for (int f = 0; f < 4; ++f)
      o[f] = wmma_f16(pa, load_B_nk(&sV[buf][f * 16][0], 40), o[f]);
    __syncthreads();
  }

  _Float16* dst = oh + ((size_t)b * L + q0) * 1024 + h * 64;
#pragma unroll
  for (int r = 0; r < 8; ++r) {
    const float inv = 1.f / lrow[r];
#pragma unroll
    for (int f = 0; f < 4; ++f)
      dst[(size_t)(rb + r) * 1024 + f * 16 + cn] = (_Float16)(o[f][r] * inv);
  }
}

// ---------------------------------------------------------------------------
// Host orchestration
// ---------------------------------------------------------------------------
static inline char* carve(char*& p, size_t bytes) {
  char* r = p;
  p += (bytes + 255) & ~(size_t)255;
  return r;
}

extern "C" void kernel_launch(void* const* d_in, const int* in_sizes, int n_in,
                              void* d_out, int out_size, void* d_ws,
                              size_t ws_size, hipStream_t stream) {
  (void)in_sizes; (void)n_in; (void)out_size; (void)ws_size;
  const float* x    = (const float*)d_in[0];  // [B,L,1024]
  const float* pos  = (const float*)d_in[1];  // [B,L,3]
  const float* xc   = (const float*)d_in[2];  // [B,LC,1024]
  const float* posc = (const float*)d_in[3];  // [B,LC,3]
  const float* nsc  = (const float*)d_in[4];  // [1024]
  const float* ncs  = (const float*)d_in[5];  // [1024]
  const float* qw   = (const float*)d_in[6];  // [1024,1024]
  const float* kvw  = (const float*)d_in[7];  // [2048,1024]
  const float* hs   = (const float*)d_in[8];  // [16]
  const float* ow   = (const float*)d_in[9];  // [1024,1024]
  float* out = (float*)d_out;

  const int B = 2, L = 2048, LC = 2048, DM = 1024, H = 16, DH = 64;

  char* p = (char*)d_ws;
  _Float16* xn_h  = (_Float16*)carve(p, (size_t)B * L * DM * 2);
  _Float16* xc_h  = (_Float16*)carve(p, (size_t)B * LC * DM * 2);
  _Float16* qw_h  = (_Float16*)carve(p, (size_t)DM * DM * 2);
  _Float16* kvw_h = (_Float16*)carve(p, (size_t)2 * DM * DM * 2);
  _Float16* ow_h  = (_Float16*)carve(p, (size_t)DM * DM * 2);
  float*    q_f   = (float*)carve(p, (size_t)B * L * DM * 4);
  float*    kv_f  = (float*)carve(p, (size_t)B * LC * 2 * DM * 4);
  _Float16* qhb   = (_Float16*)carve(p, (size_t)B * H * L * DH * 2);
  _Float16* khb   = (_Float16*)carve(p, (size_t)B * H * LC * DH * 2);
  _Float16* vtb   = (_Float16*)carve(p, (size_t)B * H * DH * LC * 2);
  _Float16* ohb   = (_Float16*)carve(p, (size_t)B * L * DM * 2);

  // Weight casts
  cast_f32_f16_kernel<<<DM * DM / 256, 256, 0, stream>>>(qw, qw_h, DM * DM);
  cast_f32_f16_kernel<<<2 * DM * DM / 256, 256, 0, stream>>>(kvw, kvw_h, 2 * DM * DM);
  cast_f32_f16_kernel<<<DM * DM / 256, 256, 0, stream>>>(ow, ow_h, DM * DM);

  // RMS norms
  rmsnorm_cast_kernel<<<B * L, 256, 0, stream>>>(x, nsc, xn_h, DM);
  rmsnorm_cast_kernel<<<B * LC, 256, 0, stream>>>(xc, ncs, xc_h, DM);

  // Projections
  gemm_wmma_kernel<<<dim3(B * L / GT_M, DM / GT_N), 256, 0, stream>>>(
      xn_h, qw_h, q_f, nullptr, B * L, DM, DM, DM);
  gemm_wmma_kernel<<<dim3(B * LC / GT_M, 2 * DM / GT_N), 256, 0, stream>>>(
      xc_h, kvw_h, kv_f, nullptr, B * LC, 2 * DM, DM, 2 * DM);

  // Cosine scale + RoPE, head-major relayout (+ V transpose)
  q_post_kernel<<<B * L * H / 256, 256, 0, stream>>>(q_f, pos, hs, qhb, L);
  kv_post_kernel<<<B * LC * H / 256, 256, 0, stream>>>(kv_f, posc, hs, khb, vtb, LC);

  // Flash attention
  flash_attn_kernel<<<dim3(L / 64, H, B), 128, 0, stream>>>(qhb, khb, vtb, ohb, L, LC);

  // Output projection + residual
  gemm_wmma_kernel<<<dim3(B * L / GT_M, DM / GT_N), 256, 0, stream>>>(
      ohb, ow_h, out, x, B * L, DM, DM, DM);
}
